// MultiHeadAttention_53352083751187
// MI455X (gfx1250) — matrix operand order repaired and checked by reading, not compiled, not run on hardware
//
#include <hip/hip_runtime.h>
#include <hip/hip_bf16.h>
#include <math.h>

// Problem constants (B=2, T=2048, C=1024, H=16, D=64)
#define TT    2048
#define CC    1024
#define NHEAD 16
#define HD    64
#define N_QKV 3072

typedef __attribute__((ext_vector_type(16))) __bf16 bf16x16;
typedef __attribute__((ext_vector_type(4)))  __bf16 bf16x4;
typedef __attribute__((ext_vector_type(8)))  float  f32x8;
typedef __attribute__((ext_vector_type(4)))  float  f32x4;
typedef __attribute__((ext_vector_type(4)))  unsigned int u32x4;

union FragB16 { bf16x16 v; u32x4 q[2]; __bf16 h[16]; };
union AccF    { f32x8 v;   float  f[8]; };

__device__ __forceinline__ f32x8 wmma_bf16f32(bf16x16 a, bf16x16 b, f32x8 c) {
  // v_wmma_f32_16x16x32_bf16  D = A x B + C  (f32 accumulate)
  return __builtin_amdgcn_wmma_f32_16x16x32_bf16(false, a, false, b, (short)0, c,
                                                 false, false);
}
// swap lanes L <-> L^16 inside wave32 (ds_swizzle group-of-32, xor=0x10, and=0x1f)
__device__ __forceinline__ float swap_half(float x) {
  return __int_as_float(__builtin_amdgcn_ds_swizzle(__float_as_int(x), 0x401F));
}
// broadcast value from lane `src`
__device__ __forceinline__ float bcast(float x, int src) {
  return __int_as_float(__builtin_amdgcn_ds_bpermute(src << 2, __float_as_int(x)));
}
// A-fragment (16x32 bf16): lane<16 holds K 0..7 & 16..23 ; lane>=16 holds K 8..15 & 24..31
__device__ __forceinline__ void load_fragA(FragB16& f, const __bf16* rowPtr,
                                           int colBase, int lane) {
  int c = colBase + ((lane & 16) ? 8 : 0);
  f.q[0] = *(const u32x4*)(rowPtr + c);
  f.q[1] = *(const u32x4*)(rowPtr + c + 16);
}
// B-fragment (32x16 bf16): lane<16 holds K 0..7 & 16..23 ; lane>=16 holds K 8..15 & 24..31
__device__ __forceinline__ void load_fragB(FragB16& f, const __bf16* rowPtr,
                                           int colBase, int lane) {
  int c = colBase + ((lane & 16) ? 8 : 0);
  f.q[0] = *(const u32x4*)(rowPtr + c);
  f.q[1] = *(const u32x4*)(rowPtr + c + 16);
}
// CDNA5 async copy: 16 bytes global -> LDS, tracked by ASYNCcnt
__device__ __forceinline__ void async_b128(unsigned ldsDst, const float* gsrc) {
  asm volatile("global_load_async_to_lds_b128 %0, %1, off"
               :: "v"(ldsDst), "v"(gsrc) : "memory");
}
__device__ __forceinline__ unsigned lds_off(const void* p) {
  // LDS aperture keeps the byte offset in addr[31:0] (ISA 10.2 aperture rules)
  return (unsigned)(size_t)p;
}

// ---------------------------------------------------------------------------
// Prep A: elementwise fp32 -> bf16 (for x). 4 elements / thread, b128 in, b64 out.
// ---------------------------------------------------------------------------
__global__ __launch_bounds__(256) void cvt_bf16(const float* __restrict__ in,
                                                __bf16* __restrict__ out) {
  const size_t i = ((size_t)blockIdx.x * blockDim.x + threadIdx.x) << 2;
  f32x4 v = *(const f32x4*)(in + i);
  bf16x4 o;
#pragma unroll
  for (int j = 0; j < 4; ++j) o[j] = (__bf16)v[j];
  *(bf16x4*)(out + i) = o;
}

// ---------------------------------------------------------------------------
// Prep B: fp32 [K][N] -> bf16 [N][K] transpose via LDS 32x32 tile (padded).
// Both global sides coalesced; GEMM B-fragments become contiguous b128 loads.
// ---------------------------------------------------------------------------
__global__ __launch_bounds__(256) void wtrans(const float* __restrict__ in,
                                              __bf16* __restrict__ outT,
                                              int K, int N) {
  __shared__ __bf16 tile[32][33];
  const int tx = threadIdx.x & 31;
  const int ty = threadIdx.x >> 5;          // 0..7
  const int n0 = blockIdx.x << 5;
  const int k0 = blockIdx.y << 5;
#pragma unroll
  for (int i = 0; i < 4; ++i) {
    const int k = k0 + ty + (i << 3);
    tile[ty + (i << 3)][tx] = (__bf16)in[(size_t)k * N + n0 + tx];
  }
  __syncthreads();
#pragma unroll
  for (int i = 0; i < 4; ++i) {
    const int n = n0 + ty + (i << 3);
    outT[(size_t)n * K + k0 + tx] = tile[tx][ty + (i << 3)];
  }
}

// ---------------------------------------------------------------------------
// Kernel 1: qkv = x @ w_qkv  (all-bf16 operands, f32 accum).
// Wave -> 32x64 tile: 2 A-frags x 4 B-frags -> 8 WMMAs per 32-deep k-step.
// Writes Q,K [B,H,T,D] and V transposed [B,H,D,T].
// ---------------------------------------------------------------------------
__global__ __launch_bounds__(256) void qkv_gemm(const __bf16* __restrict__ xbf,
                                                const __bf16* __restrict__ wT,
                                                __bf16* __restrict__ Qb,
                                                __bf16* __restrict__ Kb,
                                                __bf16* __restrict__ Vt) {
  const int lane  = threadIdx.x & 31;
  const int wid   = (blockIdx.x * blockDim.x + threadIdx.x) >> 5;
  const int mTile = wid & 127;          // 128 row tiles of 32
  const int nGrp  = wid >> 7;           // 48 col groups of 64 (block shares nGrp)
  const int m0 = mTile << 5;
  const int n0 = nGrp << 6;
  const int half8 = (lane & 16) ? 8 : 0;
  const __bf16* arow0 = xbf + ((size_t)(m0 + (lane & 15)) << 10);
  const __bf16* arow1 = arow0 + (16 << 10);

  AccF acc[2][4];
#pragma unroll
  for (int s = 0; s < 2; ++s)
#pragma unroll
    for (int i = 0; i < 4; ++i)
#pragma unroll
      for (int j = 0; j < 8; ++j) acc[s][i].f[j] = 0.0f;

  for (int kk = 0; kk < CC; kk += 32) {
    FragB16 a0, a1;
    load_fragA(a0, arow0, kk, lane);
    load_fragA(a1, arow1, kk, lane);
#pragma unroll
    for (int nt = 0; nt < 4; ++nt) {
      FragB16 bfr;
      const __bf16* brow = wT + ((size_t)(n0 + (nt << 4) + (lane & 15)) << 10);
      load_fragB(bfr, brow, kk, lane);
      acc[0][nt].v = wmma_bf16f32(a0.v, bfr.v, acc[0][nt].v);
      acc[1][nt].v = wmma_bf16f32(a1.v, bfr.v, acc[1][nt].v);
    }
  }
  // This wave's 64-column group lies entirely inside one (q/k/v, head)
  const int s  = n0 >> 10;              // 0=Q 1=K 2=V
  const int hh = (n0 & 1023) >> 6;      // head
  if (s < 2) {
    __bf16* dst = (s == 0) ? Qb : Kb;   // [B,H,T,D]: row-major 64-elem rows
#pragma unroll
    for (int sub = 0; sub < 2; ++sub) {
#pragma unroll
      for (int r = 0; r < 8; ++r) {
        const int row = m0 + (sub << 4) + r + half8;   // row in [0, B*T)
        const int b = row >> 11, t = row & (TT - 1);
        __bf16* o = dst + ((size_t)((b * NHEAD + hh) * TT + t) << 6) + (lane & 15);
#pragma unroll
        for (int nt = 0; nt < 4; ++nt) o[nt << 4] = (__bf16)acc[sub][nt].f[r];
      }
    }
  } else {
#pragma unroll
    for (int sub = 0; sub < 2; ++sub) {
#pragma unroll
      for (int nt = 0; nt < 4; ++nt) {
        const int d = (nt << 4) + (lane & 15);
#pragma unroll
        for (int r = 0; r < 8; ++r) {
          const int row = m0 + (sub << 4) + r + half8;
          const int b = row >> 11, t = row & (TT - 1);
          Vt[((size_t)((b * NHEAD + hh) * HD + d) << 11) + t] =
              (__bf16)acc[sub][nt].f[r];
        }
      }
    }
  }
}

// ---------------------------------------------------------------------------
// Kernel 2: flash attention. One wave per (b,h,16-query tile).
// Scores computed TRANSPOSED (St = K x Q^T) so St's C-layout (kj in VGPRs,
// qi in lanes) is exactly the A-fragment layout needed for P x V.
// The additive mask tile is staged via async global->LDS copies (ASYNCcnt),
// double-buffered so the copy overlaps the WMMA stream.
// ---------------------------------------------------------------------------
__global__ __launch_bounds__(256) void attn(const __bf16* __restrict__ Qb,
                                            const __bf16* __restrict__ Kb,
                                            const __bf16* __restrict__ Vt,
                                            const float* __restrict__ mask,
                                            const float* __restrict__ head_mask,
                                            __bf16* __restrict__ ybf) {
  __shared__ float msk[8][2][16][32];   // [wave][buf][q-row][key-col] = 32 KB
  const int lane  = threadIdx.x & 31;
  const int wv    = threadIdx.x >> 5;   // wave in block
  const int wid   = (blockIdx.x * blockDim.x + threadIdx.x) >> 5;
  const int qTile = wid & 127;          // 128 q-tiles, block shares (b,h)
  const int bh    = wid >> 7;           // 0..31
  const int hh    = bh & (NHEAD - 1);
  const int b     = bh >> 4;
  const int qt0   = qTile << 4;
  const int half8 = (lane & 16) ? 8 : 0;
  const int tq    = qt0 + (lane & 15);
  const float NEGINF = -__builtin_inff();

  const __bf16* Qh = Qb + ((size_t)bh << 17);    // bh * T * D
  const __bf16* Kh = Kb + ((size_t)bh << 17);
  const __bf16* Vh = Vt + ((size_t)bh << 17);
  const float* mrow = mask + ((size_t)b * TT + tq) * TT;   // this lane's q-row

  // LDS byte addresses of this lane's mask row in each buffer
  const unsigned lrow0 = lds_off(&msk[wv][0][lane & 15][0]);
  const unsigned lrow1 = lds_off(&msk[wv][1][lane & 15][0]);

  // Resident Q B-fragments (16 queries x 64 d) : qf0 = d 0..31, qf1 = d 32..63
  FragB16 qf0, qf1;
  const __bf16* qrow = Qh + ((size_t)(qt0 + (lane & 15)) << 6);
  load_fragB(qf0, qrow, 0, lane);
  load_fragB(qf1, qrow, 32, lane);

  AccF y[4];
#pragma unroll
  for (int i = 0; i < 4; ++i)
#pragma unroll
    for (int j = 0; j < 8; ++j) y[i].f[j] = 0.0f;

  float m = NEGINF, l = 0.0f;
  int ktEnd = ((qt0 + 16 + 31) >> 5) << 5;       // causal: keys <= qt0+15
  if (ktEnd > TT) ktEnd = TT;                    // (last tile covers all of T)

  // Preload mask tile for kt=0 into buffer 0 (4 async b128 per lane = 64 B)
  {
    const float* g = mrow + half8;
    async_b128(lrow0 + ((half8 + 0) << 2),  g + 0);
    async_b128(lrow0 + ((half8 + 4) << 2),  g + 4);
    async_b128(lrow0 + ((half8 + 16) << 2), g + 16);
    async_b128(lrow0 + ((half8 + 20) << 2), g + 20);
  }
  int buf = 0;

  for (int kt = 0; kt < ktEnd; kt += 32) {
    // K A-fragments: tile0 = keys kt..kt+15, tile1 = kt+16..kt+31, d split 0/32
    const __bf16* krow0 = Kh + ((size_t)(kt + (lane & 15)) << 6);
    const __bf16* krow1 = krow0 + (16 << 6);
    __builtin_prefetch((const void*)(krow0 + (32 << 6)), 0, 1);
    FragB16 k00, k01, k10, k11;
    load_fragA(k00, krow0, 0, lane);
    load_fragA(k01, krow0, 32, lane);
    load_fragA(k10, krow1, 0, lane);
    load_fragA(k11, krow1, 32, lane);

    AccF s0, s1;
#pragma unroll
    for (int j = 0; j < 8; ++j) { s0.f[j] = 0.0f; s1.f[j] = 0.0f; }
    s0.v = wmma_bf16f32(k00.v, qf0.v, s0.v);
    s0.v = wmma_bf16f32(k01.v, qf1.v, s0.v);
    s1.v = wmma_bf16f32(k10.v, qf0.v, s1.v);
    s1.v = wmma_bf16f32(k11.v, qf1.v, s1.v);

    // Kick off async mask copy for the NEXT key block, then wait for current.
    if (kt + 32 < ktEnd) {
      const unsigned lnext = buf ? lrow0 : lrow1;
      const float* g = mrow + (kt + 32) + half8;
      async_b128(lnext + ((half8 + 0) << 2),  g + 0);
      async_b128(lnext + ((half8 + 4) << 2),  g + 4);
      async_b128(lnext + ((half8 + 16) << 2), g + 16);
      async_b128(lnext + ((half8 + 20) << 2), g + 20);
      asm volatile("s_wait_asynccnt 4" ::: "memory");   // current 4 done (in-order)
    } else {
      asm volatile("s_wait_asynccnt 0" ::: "memory");
    }

    // scale + causal + additive mask.  St element r <-> key tk0+r (tile0) / +16
    const float* ml = &msk[wv][buf][lane & 15][half8];
    f32x4 ma0 = *(const f32x4*)(ml);
    f32x4 ma1 = *(const f32x4*)(ml + 4);
    f32x4 mb0 = *(const f32x4*)(ml + 16);
    f32x4 mb1 = *(const f32x4*)(ml + 20);
    const int tk0 = kt + half8;
    float sv0[8], sv1[8];
    float mx = NEGINF;
#pragma unroll
    for (int r = 0; r < 8; ++r) {
      const int tk = tk0 + r;
      float v0 = s0.f[r] * 0.125f + ((r < 4) ? ma0[r] : ma1[r - 4]);
      float v1 = s1.f[r] * 0.125f + ((r < 4) ? mb0[r] : mb1[r - 4]);
      if ((tk > tq) && (tq != TT - 1)) v0 = NEGINF;
      if ((tk + 16 > tq) && (tq != TT - 1)) v1 = NEGINF;
      sv0[r] = v0; sv1[r] = v1;
      mx = fmaxf(mx, fmaxf(v0, v1));
    }
    buf ^= 1;
    mx = fmaxf(mx, swap_half(mx));                 // reduce over both lane halves
    const float mnew  = fmaxf(m, mx);
    const float alpha = __expf(m - mnew);

    FragB16 p;
    float psum = 0.0f;
#pragma unroll
    for (int r = 0; r < 8; ++r) {
      float p0 = __expf(sv0[r] - mnew);
      float p1 = __expf(sv1[r] - mnew);
      p.h[r]     = (__bf16)p0;                      // kj local r (+half8)
      p.h[8 + r] = (__bf16)p1;                      // kj local 16+r (+half8)
      psum += p0 + p1;
    }
    psum += swap_half(psum);
    l = l * alpha + psum;
    m = mnew;

    // rescale Y accumulators: row qi lives in VGPR dim -> broadcast alpha[qi]
#pragma unroll
    for (int r = 0; r < 8; ++r) {
      const float av = bcast(alpha, r + half8);
      y[0].f[r] *= av; y[1].f[r] *= av; y[2].f[r] *= av; y[3].f[r] *= av;
    }
    // P x V : V stored transposed [d][t] -> contiguous B-fragments
#pragma unroll
    for (int dt = 0; dt < 4; ++dt) {
      FragB16 vf;
      const __bf16* vrow = Vh + ((size_t)((dt << 4) + (lane & 15)) << 11);
      load_fragB(vf, vrow, kt, lane);
      y[dt].v = wmma_bf16f32(p.v, vf.v, y[dt].v);
    }
  }

  const float hm  = head_mask[hh];
  const float inv = hm / l;                         // per lane: row qi = lane&15
#pragma unroll
  for (int r = 0; r < 8; ++r) {
    const float sc = bcast(inv, r + half8);
    const int t = qt0 + r + half8;
#pragma unroll
    for (int dt = 0; dt < 4; ++dt) {
      const int d = (dt << 4) + (lane & 15);
      ybf[((size_t)(b * TT + t) << 10) + hh * HD + d] = (__bf16)(y[dt].f[r] * sc);
    }
  }
}

// ---------------------------------------------------------------------------
// Kernel 3: out = y @ w_proj  (bf16 x bf16 -> fp32). Wave -> 32x64 tile.
// ---------------------------------------------------------------------------
__global__ __launch_bounds__(256) void proj_gemm(const __bf16* __restrict__ ybf,
                                                 const __bf16* __restrict__ wT,
                                                 float* __restrict__ out) {
  const int lane  = threadIdx.x & 31;
  const int wid   = (blockIdx.x * blockDim.x + threadIdx.x) >> 5;
  const int mTile = wid & 127;          // 128 row tiles of 32
  const int nGrp  = wid >> 7;           // 16 col groups of 64
  const int m0 = mTile << 5;
  const int n0 = nGrp << 6;
  const int half8 = (lane & 16) ? 8 : 0;
  const __bf16* arow0 = ybf + ((size_t)(m0 + (lane & 15)) << 10);
  const __bf16* arow1 = arow0 + (16 << 10);

  AccF acc[2][4];
#pragma unroll
  for (int s = 0; s < 2; ++s)
#pragma unroll
    for (int i = 0; i < 4; ++i)
#pragma unroll
      for (int j = 0; j < 8; ++j) acc[s][i].f[j] = 0.0f;

  for (int kk = 0; kk < CC; kk += 32) {
    FragB16 a0, a1;
    load_fragA(a0, arow0, kk, lane);
    load_fragA(a1, arow1, kk, lane);
#pragma unroll
    for (int nt = 0; nt < 4; ++nt) {
      FragB16 bfr;
      const __bf16* brow = wT + ((size_t)(n0 + (nt << 4) + (lane & 15)) << 10);
      load_fragB(bfr, brow, kk, lane);
      acc[0][nt].v = wmma_bf16f32(a0.v, bfr.v, acc[0][nt].v);
      acc[1][nt].v = wmma_bf16f32(a1.v, bfr.v, acc[1][nt].v);
    }
  }
#pragma unroll
  for (int sub = 0; sub < 2; ++sub) {
#pragma unroll
    for (int r = 0; r < 8; ++r) {
      const int row = m0 + (sub << 4) + r + half8;
      float* o = out + (size_t)row * CC + n0 + (lane & 15);
#pragma unroll
      for (int nt = 0; nt < 4; ++nt) o[nt << 4] = acc[sub][nt].f[r];
    }
  }
}

// ---------------------------------------------------------------------------
extern "C" void kernel_launch(void* const* d_in, const int* in_sizes, int n_in,
                              void* d_out, int out_size, void* d_ws, size_t ws_size,
                              hipStream_t stream) {
  const float* x         = (const float*)d_in[0];   // (B,T,C)
  const float* mask      = (const float*)d_in[1];   // (B,1,T,T)
  const float* w_qkv     = (const float*)d_in[2];   // (C,3C)
  const float* w_proj    = (const float*)d_in[3];   // (C,C)
  const float* head_mask = (const float*)d_in[4];   // (H,)

  char* ws = (char*)d_ws;
  const size_t MB = (size_t)1024 * 1024;
  __bf16* xbf    = (__bf16*)(ws + 0 * MB);    //  8 MB: [B*T][C] bf16
  __bf16* WqkvT  = (__bf16*)(ws + 8 * MB);    //  6 MB: [3C][C] bf16 (transposed)
  __bf16* WprojT = (__bf16*)(ws + 14 * MB);   //  2 MB: [C][C]  bf16 (transposed)
  __bf16* Qb     = (__bf16*)(ws + 16 * MB);   //  8 MB: [B,H,T,D]
  __bf16* Kb     = (__bf16*)(ws + 24 * MB);   //  8 MB: [B,H,T,D]
  __bf16* Vt     = (__bf16*)(ws + 32 * MB);   //  8 MB: [B,H,D,T] (transposed)
  __bf16* ybf    = (__bf16*)(ws + 40 * MB);   //  8 MB: [B,T,C]

  // Prep: x -> bf16 (4M elems / 4 per thread), weights -> bf16 transposed
  cvt_bf16<<<4096, 256, 0, stream>>>(x, xbf);
  wtrans<<<dim3(N_QKV / 32, CC / 32), 256, 0, stream>>>(w_qkv, WqkvT, CC, N_QKV);
  wtrans<<<dim3(CC / 32, CC / 32), 256, 0, stream>>>(w_proj, WprojT, CC, CC);

  // 128*48 waves / 8 per block
  qkv_gemm<<<768, 256, 0, stream>>>(xbf, WqkvT, Qb, Kb, Vt);
  // 2*16*128 waves / 8 per block
  attn<<<512, 256, 0, stream>>>(Qb, Kb, Vt, mask, head_mask, ybf);
  // 128*16 waves / 8 per block
  proj_gemm<<<256, 256, 0, stream>>>(ybf, WprojT, (float*)d_out);
}
